// SimpleGRU_89644557402950
// MI455X (gfx1250) — compile-verified, matrix operand order checked by reading, not run on hardware
//
#include <hip/hip_runtime.h>

// GRU dims (fixed by the reference)
#define BATCH 128
#define TLEN  512
#define IDIM  256
#define HDIM  512
#define ODIM  256

typedef __attribute__((ext_vector_type(16))) _Float16 v16h;
typedef __attribute__((ext_vector_type(8)))  float    v8f;

union Frag16 { v16h v; uint4 q[2]; };
union Pack16 { _Float16 h[16]; uint4 q[2]; };

#define LDS_STRIDE 136   // 128 + 8 pad (halfwords) -> conflict-free b128 reads
#define KC 128           // K-chunk staged per barrier

__device__ __forceinline__ float sigmoidf_(float x) {
    return 1.0f / (1.0f + __expf(-x));
}

// Stage a 32-row x 128-col f32 tile into LDS as f16. 256 threads: 8 threads/row,
// 16 contiguous cols each (4x float4 loads -> 2x uint4 LDS stores).
__device__ __forceinline__ void stage_tile(_Float16* lds, const float* __restrict__ src,
                                           int row0, int ldsrc, int kc) {
    const int t  = threadIdx.x;      // 0..255
    const int r  = t >> 3;           // 0..31
    const int c0 = (t & 7) * 16;     // 0,16,...,112
    const float4* p4 = (const float4*)(src + (size_t)(row0 + r) * (size_t)ldsrc + kc + c0);
    Pack16 pk;
#pragma unroll
    for (int q = 0; q < 4; ++q) {
        float4 f = p4[q];
        pk.h[q * 4 + 0] = (_Float16)f.x;
        pk.h[q * 4 + 1] = (_Float16)f.y;
        pk.h[q * 4 + 2] = (_Float16)f.z;
        pk.h[q * 4 + 3] = (_Float16)f.w;
    }
    *(uint4*)(&lds[r * LDS_STRIDE + c0])     = pk.q[0];
    *(uint4*)(&lds[r * LDS_STRIDE + c0 + 8]) = pk.q[1];
}

// A fragment (16x32 f16, M = lane&15): elements 0..7 = K[ks+kh8 .. +7],
// elements 8..15 = K[ks+kh8+16 .. +23], kh8 = 8 if lane>=16.
__device__ __forceinline__ v16h load_a_lds(const _Float16* lds, int rowInTile, int ks, int kh8) {
    const _Float16* p = lds + rowInTile * LDS_STRIDE + ks + kh8;
    Frag16 f;
    f.q[0] = *(const uint4*)(p);
    f.q[1] = *(const uint4*)(p + 16);
    return f.v;
}

// B fragment (32x16 f16, N = lane&15): same K split, sourced from f16 weight
// row `row` of length K (row-major [rows, K] => implicit transpose for W^T).
__device__ __forceinline__ v16h load_b_gl(const _Float16* __restrict__ W, size_t row, int K, int k, int kh8) {
    const _Float16* p = W + row * (size_t)K + k + kh8;
    Frag16 f;
    f.q[0] = *(const uint4*)(p);
    f.q[1] = *(const uint4*)(p + 16);
    return f.v;
}

__global__ void __launch_bounds__(256)
f32_to_f16_kernel(const float* __restrict__ src, _Float16* __restrict__ dst, int n) {
    int i = blockIdx.x * 256 + threadIdx.x;
    if (i < n) dst[i] = (_Float16)src[i];
}

// One GRU cell: hout = GRUCell(inp, hin). Block computes 32(B) x 64(H) tile;
// 8 waves arranged 2(B) x 4(H), one 16x16 WMMA subtile per wave.
__global__ void __launch_bounds__(256)
gru_cell_kernel(const float* __restrict__ inp, int ldInp, int Ki,
                const float* __restrict__ hin,
                const _Float16* __restrict__ Wih,   // [3H, Ki] f16, row-major
                const _Float16* __restrict__ Whh,   // [3H, H]  f16, row-major
                const float* __restrict__ b_ih,
                const float* __restrict__ b_hh,
                float* __restrict__ hout) {
    __shared__ _Float16 lds[32 * LDS_STRIDE];

    const int b0   = blockIdx.x * 32;
    const int j0   = blockIdx.y * 64;
    const int lane = threadIdx.x & 31;
    const int wave = threadIdx.x >> 5;
    const int wb   = wave >> 2;                 // 0..1  (B subtile)
    const int wj   = wave & 3;                  // 0..3  (H subtile)
    const int kh8  = ((lane >> 4) & 1) * 8;     // K half split per lane group
    const int arow = wb * 16 + (lane & 15);     // A row within LDS tile
    const int j    = j0 + wj * 16 + (lane & 15);// gate-unit index / B row

    v8f acc_r = {}, acc_z = {}, acc_in = {}, acc_hn = {};

    // gi = inp @ Wih^T  (gates r,z,n accumulate from input path)
    for (int kc = 0; kc < Ki; kc += KC) {
        __syncthreads();
        stage_tile(lds, inp, b0, ldInp, kc);
        __syncthreads();
#pragma unroll
        for (int ks = 0; ks < KC; ks += 32) {
            v16h a  = load_a_lds(lds, arow, ks, kh8);
            v16h br = load_b_gl(Wih, (size_t)(0 * HDIM + j), Ki, kc + ks, kh8);
            v16h bz = load_b_gl(Wih, (size_t)(1 * HDIM + j), Ki, kc + ks, kh8);
            v16h bn = load_b_gl(Wih, (size_t)(2 * HDIM + j), Ki, kc + ks, kh8);
            acc_r  = __builtin_amdgcn_wmma_f32_16x16x32_f16(false, a, false, br, (short)0, acc_r,  false, false);
            acc_z  = __builtin_amdgcn_wmma_f32_16x16x32_f16(false, a, false, bz, (short)0, acc_z,  false, false);
            acc_in = __builtin_amdgcn_wmma_f32_16x16x32_f16(false, a, false, bn, (short)0, acc_in, false, false);
        }
    }

    // gh = hin @ Whh^T  (r,z fuse into same accumulators; n-path kept separate)
    for (int kc = 0; kc < HDIM; kc += KC) {
        __syncthreads();
        stage_tile(lds, hin, b0, HDIM, kc);
        __syncthreads();
#pragma unroll
        for (int ks = 0; ks < KC; ks += 32) {
            v16h a  = load_a_lds(lds, arow, ks, kh8);
            v16h br = load_b_gl(Whh, (size_t)(0 * HDIM + j), HDIM, kc + ks, kh8);
            v16h bz = load_b_gl(Whh, (size_t)(1 * HDIM + j), HDIM, kc + ks, kh8);
            v16h bn = load_b_gl(Whh, (size_t)(2 * HDIM + j), HDIM, kc + ks, kh8);
            acc_r  = __builtin_amdgcn_wmma_f32_16x16x32_f16(false, a, false, br, (short)0, acc_r,  false, false);
            acc_z  = __builtin_amdgcn_wmma_f32_16x16x32_f16(false, a, false, bz, (short)0, acc_z,  false, false);
            acc_hn = __builtin_amdgcn_wmma_f32_16x16x32_f16(false, a, false, bn, (short)0, acc_hn, false, false);
        }
    }

    // Fused epilogue on the C/D layout: m = v + 8*(lane>=16), n = lane&15.
    const float bir = b_ih[j],            bhr = b_hh[j];
    const float biz = b_ih[HDIM + j],     bhz = b_hh[HDIM + j];
    const float bin = b_ih[2 * HDIM + j], bhn = b_hh[2 * HDIM + j];
#pragma unroll
    for (int v = 0; v < 8; ++v) {
        const int m = v + kh8;                  // kh8 is 0 or 8 == row offset
        const int b = b0 + wb * 16 + m;
        const float r  = sigmoidf_(acc_r[v] + bir + bhr);
        const float z  = sigmoidf_(acc_z[v] + biz + bhz);
        const float nn = tanhf(acc_in[v] + bin + r * (acc_hn[v] + bhn));
        const float hp = hin[(size_t)b * HDIM + j];
        hout[(size_t)b * HDIM + j] = (1.0f - z) * nn + z * hp;
    }
}

// y_t = h @ Wout^T + b_out ; block = 32(B) x 64(O) tile, same fragment scheme.
__global__ void __launch_bounds__(256)
gru_out_kernel(const float* __restrict__ h,
               const _Float16* __restrict__ Wout,  // [O, H] f16 row-major
               const float* __restrict__ b_out,
               float* __restrict__ y, int ldY) {   // y pre-offset to step t
    __shared__ _Float16 lds[32 * LDS_STRIDE];

    const int b0   = blockIdx.x * 32;
    const int o0   = blockIdx.y * 64;
    const int lane = threadIdx.x & 31;
    const int wave = threadIdx.x >> 5;
    const int wb   = wave >> 2;
    const int wo   = wave & 3;
    const int kh8  = ((lane >> 4) & 1) * 8;
    const int arow = wb * 16 + (lane & 15);
    const int o    = o0 + wo * 16 + (lane & 15);

    v8f acc = {};
    for (int kc = 0; kc < HDIM; kc += KC) {
        __syncthreads();
        stage_tile(lds, h, b0, HDIM, kc);
        __syncthreads();
#pragma unroll
        for (int ks = 0; ks < KC; ks += 32) {
            v16h a  = load_a_lds(lds, arow, ks, kh8);
            v16h bo = load_b_gl(Wout, (size_t)o, HDIM, kc + ks, kh8);
            acc = __builtin_amdgcn_wmma_f32_16x16x32_f16(false, a, false, bo, (short)0, acc, false, false);
        }
    }
    const float bias = b_out[o];
#pragma unroll
    for (int v = 0; v < 8; ++v) {
        const int b = b0 + wb * 16 + v + kh8;
        y[(size_t)b * ldY + o] = acc[v] + bias;
    }
}

extern "C" void kernel_launch(void* const* d_in, const int* in_sizes, int n_in,
                              void* d_out, int out_size, void* d_ws, size_t ws_size,
                              hipStream_t stream) {
    (void)in_sizes; (void)n_in; (void)out_size; (void)ws_size;

    const float* x      = (const float*)d_in[0];   // [B,T,I]
    const float* h0     = (const float*)d_in[1];   // [B,H]
    const float* W_ih0  = (const float*)d_in[2];   // [3H,I]
    const float* W_hh0  = (const float*)d_in[3];   // [3H,H]
    const float* b_ih0  = (const float*)d_in[4];
    const float* b_hh0  = (const float*)d_in[5];
    const float* W_ih1  = (const float*)d_in[6];   // [3H,H]
    const float* W_hh1  = (const float*)d_in[7];   // [3H,H]
    const float* b_ih1  = (const float*)d_in[8];
    const float* b_hh1  = (const float*)d_in[9];
    const float* W_out  = (const float*)d_in[10];  // [O,H]
    const float* b_out  = (const float*)d_in[11];
    float* Y = (float*)d_out;                      // [B,T,O]

    // Workspace carve-up (256B-aligned): f16 weights + f32 h ping-pong (~6.3 MB).
    char*  ws  = (char*)d_ws;
    size_t off = 0;
    auto carve = [&](size_t bytes) -> void* {
        void* p = ws + off;
        off = (off + bytes + 255) & ~(size_t)255;
        return p;
    };
    _Float16* wih0 = (_Float16*)carve((size_t)3 * HDIM * IDIM * 2);
    _Float16* whh0 = (_Float16*)carve((size_t)3 * HDIM * HDIM * 2);
    _Float16* wih1 = (_Float16*)carve((size_t)3 * HDIM * HDIM * 2);
    _Float16* whh1 = (_Float16*)carve((size_t)3 * HDIM * HDIM * 2);
    _Float16* wout = (_Float16*)carve((size_t)ODIM * HDIM * 2);
    float*    hA   = (float*)carve((size_t)BATCH * HDIM * 4);
    float*    hB   = (float*)carve((size_t)BATCH * HDIM * 4);

    // One-time f32 -> f16 weight conversion (L2-resident for all 512 steps).
    auto conv = [&](const float* s, _Float16* d, int n) {
        f32_to_f16_kernel<<<(n + 255) / 256, 256, 0, stream>>>(s, d, n);
    };
    conv(W_ih0, wih0, 3 * HDIM * IDIM);
    conv(W_hh0, whh0, 3 * HDIM * HDIM);
    conv(W_ih1, wih1, 3 * HDIM * HDIM);
    conv(W_hh1, whh1, 3 * HDIM * HDIM);
    conv(W_out, wout, ODIM * HDIM);

    // Seed recurrent state (capture-safe async D2D).
    hipMemcpyAsync(hA, h0, (size_t)BATCH * HDIM * sizeof(float),
                   hipMemcpyDeviceToDevice, stream);

    const dim3 cellGrid(BATCH / 32, HDIM / 64);  // 4 x 8
    const dim3 outGrid (BATCH / 32, ODIM / 64);  // 4 x 4

    for (int t = 0; t < TLEN; ++t) {
        // Layer 0: h' = cell(x_t, h)
        gru_cell_kernel<<<cellGrid, 256, 0, stream>>>(
            x + (size_t)t * IDIM, TLEN * IDIM, IDIM,
            hA, wih0, whh0, b_ih0, b_hh0, hB);
        // Layer 1 (shared-state quirk): h = cell(h', h')
        gru_cell_kernel<<<cellGrid, 256, 0, stream>>>(
            hB, HDIM, HDIM,
            hB, wih1, whh1, b_ih1, b_hh1, hA);
        // y_t = h @ Wout^T + b_out
        gru_out_kernel<<<outGrid, 256, 0, stream>>>(
            hA, wout, b_out, Y + (size_t)t * ODIM, TLEN * ODIM);
    }
}
